// unidirectional_placebo_36344013258823
// MI455X (gfx1250) — matrix operand order repaired and checked by reading, not compiled, not run on hardware
//
#include <hip/hip_runtime.h>
#include <hip/hip_bf16.h>

#define B_SZ    4
#define DM      1024
#define SEQ_L   8192
#define XPAD    256     // front zero pad for x (covers s down to -256)
#define KRPAD   16      // tail zero pad for reversed kernel (causal zeros)

typedef __attribute__((ext_vector_type(16))) __bf16         v16bf;
typedef __attribute__((ext_vector_type(16))) unsigned short v16u;
typedef __attribute__((ext_vector_type(8)))  float          v8f;
typedef __attribute__((ext_vector_type(4)))  unsigned int   v4u;
typedef __attribute__((ext_vector_type(2)))  unsigned int   v2u;

struct AFrag { v4u lo, hi; };   // 32 bytes == v16bf

__device__ __forceinline__ void sched_fence() {
#if __has_builtin(__builtin_amdgcn_sched_barrier)
  __builtin_amdgcn_sched_barrier(0);   // pin: nothing moves across
#endif
}

__device__ __forceinline__ unsigned short f2bf(float f) {
  unsigned int u = __builtin_bit_cast(unsigned int, f);
  u += 0x7FFFu + ((u >> 16) & 1u);            // round-to-nearest-even
  return (unsigned short)(u >> 16);
}
__device__ __forceinline__ unsigned int pack2(float a, float b) {
  return (unsigned int)f2bf(a) | ((unsigned int)f2bf(b) << 16);
}

// One workgroup per (b,d) row. Stage the x row (bf16) and the *time-reversed*
// combined kernel row kr[p] = (k1+k2)[8191-p] (bf16) into LDS once. 8 waves,
// each owning 4 output groups of 256 time steps, run a segmented odd-lag loop:
// per lag, ALL LDS fragments (one shared Toeplitz B + up to 4 A's) are issued
// ahead of a sched_barrier, then the independent WMMAs follow -> staggered
// s_wait_dscnt, XDL pipelining across groups. Scalar (SGPR) loop bounds keep
// EXEC all-ones as WMMA requires.
__global__ __launch_bounds__(256)
void toeplitz_wmma_causal_conv(const float* __restrict__ x,
                               const float* __restrict__ k1,
                               const float* __restrict__ k2,
                               const float* __restrict__ Dp,
                               float* __restrict__ y) {
  __shared__ __align__(16) unsigned short x_lds[XPAD + SEQ_L];    // bf16 bits
  __shared__ __align__(16) unsigned short kr_lds[SEQ_L + KRPAD];  // bf16 bits, reversed

  const int tid = threadIdx.x;        // 0..255
  const int bd  = blockIdx.x;         // b*DM + d
  const int d   = bd & (DM - 1);

  const float* xg  = x  + (size_t)bd * SEQ_L;
  const float* k1g = k1 + (size_t)d  * SEQ_L;
  const float* k2g = k2 + (size_t)d  * SEQ_L;
  float*       yg  = y  + (size_t)bd * SEQ_L;

  // Warm L2 for the kernel rows while x streams in (global_prefetch_b8).
  __builtin_prefetch(k1g + tid * 8, 0, 1);
  __builtin_prefetch(k2g + tid * 8, 0, 1);

  // Zero pads: x front pad (s < 0), kr tail pad (t-s < 0 => causal zero).
  if (tid < XPAD) x_lds[tid] = 0;
  if (tid < KRPAD) kr_lds[SEQ_L + tid] = 0;

  // Cooperative f32 -> bf16 staging: 2048 float4 loads / 256 threads.
  const float4* xg4  = reinterpret_cast<const float4*>(xg);
  const float4* k1g4 = reinterpret_cast<const float4*>(k1g);
  const float4* k2g4 = reinterpret_cast<const float4*>(k2g);
#pragma unroll 2
  for (int i = tid; i < SEQ_L / 4; i += 256) {
    float4 v = xg4[i];
    v2u px = { pack2(v.x, v.y), pack2(v.z, v.w) };
    *reinterpret_cast<v2u*>(x_lds + XPAD + i * 4) = px;
    float4 a = k1g4[i];
    float4 b = k2g4[i];
    // reversed store: kr[8188-4i .. 8191-4i] = kc[4i+3], kc[4i+2], kc[4i+1], kc[4i]
    v2u pk = { pack2(a.w + b.w, a.z + b.z), pack2(a.y + b.y, a.x + b.x) };
    *reinterpret_cast<v2u*>(kr_lds + (SEQ_L - 4 - i * 4)) = pk;
  }
  __syncthreads();

  // Scalarize the wave index: group bounds live in SGPRs -> s_cbranch loops.
  const int wave  = __builtin_amdgcn_readfirstlane(tid) >> 5;   // 0..7
  const int lane  = tid & 31;
  const int lh    = lane >> 4;        // lane half: 0 or 1
  const int n     = lane & 15;        // N column / A row index
  const int akoff = lh << 3;          // A K-offset: 0 or 8
  const int bkoff = lh << 3;          // B K-offset: 0 or 8

  const float Dval = Dp[d];

  // Groups owned by this wave: i0 = 16*(wave + 8q), q = 0..3 (increasing).
  const int i0_0 = wave << 4;
  const int i0_1 = (wave + 8) << 4;
  const int i0_2 = (wave + 16) << 4;
  const int i0_3 = (wave + 24) << 4;

  v8f acc0 = (v8f){0.f,0.f,0.f,0.f,0.f,0.f,0.f,0.f};
  v8f acc1 = acc0, acc2 = acc0, acc3 = acc0;

  // B fragment: B[kk,n] = kc[l*16 + n - kk] = kr[(8191 - l*16 - n + bkoff) + e + ((e >> 3) << 3)]
  auto bfrag_at = [&](int l) -> v16bf {
    const int rbase = (SEQ_L - 1) - (l << 4) - n + bkoff;
    v16u bu;
#pragma unroll
    for (int e = 0; e < 16; ++e) bu[e] = kr_lds[rbase + e + ((e >> 3) << 3)];
    return __builtin_bit_cast(v16bf, bu);
  };
  // A fragment: A[m,kk] = x[(i0-l+m)*16 + kk], m = n.
  auto aload = [&](int i0, int l) -> v16bf {
    const unsigned short* xa = x_lds + (XPAD + ((i0 - l + n) << 4) + akoff);
    AFrag af;
    af.lo = *reinterpret_cast<const v4u*>(xa);        // ds_load_b128
    af.hi = *reinterpret_cast<const v4u*>(xa + 16);   // ds_load_b128
    return __builtin_bit_cast(v16bf, af);
  };
  auto amac = [&](v16bf a, v16bf b, v8f& acc) {
    acc = __builtin_amdgcn_wmma_f32_16x16x32_bf16(
        false, a, false, b, (short)0, acc, false, false);
  };

  // Segmented odd-lag loop: the active-group set only shrinks as l grows.
  int l = 1;
  for (; l < i0_0 + 16; l += 2) {
    const v16bf bf = bfrag_at(l);          // issue ALL fragment loads,
    const v16bf a0 = aload(i0_0, l);       // pin them ahead of the MACs,
    const v16bf a1 = aload(i0_1, l);       // then 4 independent WMMAs:
    const v16bf a2 = aload(i0_2, l);       // staggered dscnt waits +
    const v16bf a3 = aload(i0_3, l);       // XDL pipelining across groups.
    sched_fence();
    amac(a0, bf, acc0);
    amac(a1, bf, acc1);
    amac(a2, bf, acc2);
    amac(a3, bf, acc3);
  }
  for (; l < i0_1 + 16; l += 2) {
    const v16bf bf = bfrag_at(l);
    const v16bf a1 = aload(i0_1, l);
    const v16bf a2 = aload(i0_2, l);
    const v16bf a3 = aload(i0_3, l);
    sched_fence();
    amac(a1, bf, acc1);
    amac(a2, bf, acc2);
    amac(a3, bf, acc3);
  }
  for (; l < i0_2 + 16; l += 2) {
    const v16bf bf = bfrag_at(l);
    const v16bf a2 = aload(i0_2, l);
    const v16bf a3 = aload(i0_3, l);
    sched_fence();
    amac(a2, bf, acc2);
    amac(a3, bf, acc3);
  }
  for (; l < i0_3 + 16; l += 2) {
    const v16bf bf = bfrag_at(l);
    const v16bf a3 = aload(i0_3, l);
    amac(a3, bf, acc3);
  }

  // Epilogue: y[t] = acc + x[t] * D[d]   (C/D layout: M = v + 8*lh, N = n)
  const v8f accq[4] = {acc0, acc1, acc2, acc3};
  const int i0q[4]  = {i0_0, i0_1, i0_2, i0_3};
#pragma unroll
  for (int q = 0; q < 4; ++q) {
#pragma unroll
    for (int v = 0; v < 8; ++v) {
      const int t = ((i0q[q] + v + (lh << 3)) << 4) + n;
      yg[t] = accq[q][v] + xg[t] * Dval;
    }
  }
}

extern "C" void kernel_launch(void* const* d_in, const int* in_sizes, int n_in,
                              void* d_out, int out_size, void* d_ws, size_t ws_size,
                              hipStream_t stream) {
  (void)in_sizes; (void)n_in; (void)out_size; (void)d_ws; (void)ws_size;
  const float* x  = (const float*)d_in[0];
  const float* k1 = (const float*)d_in[1];
  const float* k2 = (const float*)d_in[2];
  const float* Dp = (const float*)d_in[3];
  float* y = (float*)d_out;
  hipLaunchKernelGGL(toeplitz_wmma_causal_conv,
                     dim3(B_SZ * DM), dim3(256), 0, stream,
                     x, k1, k2, Dp, y);
}
